// Attention_2826088481156
// MI455X (gfx1250) — matrix operand order repaired and checked by reading, not compiled, not run on hardware
//
#include <hip/hip_runtime.h>

// ---------------------------------------------------------------------------
// Types
// ---------------------------------------------------------------------------
typedef __bf16 bf16_t;
typedef __attribute__((ext_vector_type(16))) __bf16 v16bf;
typedef __attribute__((ext_vector_type(8)))  float  v8f;
typedef __attribute__((ext_vector_type(4)))  unsigned int u32x4;

#define DIM   1024
#define NHEAD 16
#define HDIM  64
#define TSEQ  2048
#define BATCH 2
#define SCALE 0.125f           // 1/sqrt(64)
#define MINLOG 1e-15f

union FragBF { v16bf v; u32x4 u[2]; };

// A-matrix fragment (16x32 bf16, row-major source, leading dim ld):
// lane: m = row0 + lane%16 ; kg = lane/16 ; holds 16 contiguous K starting at kg*16
__device__ __forceinline__ v16bf load_frag_a(const bf16_t* __restrict__ p, int ld,
                                             int row0, int k0, int lane) {
  int m  = row0 + (lane & 15);
  int kg = lane >> 4;
  const bf16_t* base = p + (size_t)m * ld + k0 + kg * 16;
  FragBF f;
  f.u[0] = *(const u32x4*)(base);
  f.u[1] = *(const u32x4*)(base + 8);
  return f.v;
}

// B-matrix fragment (32x16 bf16). Source stored as B^T row-major (row n holds
// the K-contiguous column n), leading dim ld:
// lane: n = col0 + lane%16 ; holds 16 contiguous K starting at (lane/16)*16
__device__ __forceinline__ v16bf load_frag_b(const bf16_t* __restrict__ p, int ld,
                                             int col0, int k0, int lane) {
  int n  = col0 + (lane & 15);
  int kg = lane >> 4;
  const bf16_t* base = p + (size_t)n * ld + k0 + kg * 16;
  FragBF f;
  f.u[0] = *(const u32x4*)(base);
  f.u[1] = *(const u32x4*)(base + 8);
  return f.v;
}

__device__ __forceinline__ v8f wmma_bf16(v16bf a, v16bf b, v8f c) {
  return __builtin_amdgcn_wmma_f32_16x16x32_bf16(false, a, false, b,
                                                 (short)0, c, false, false);
}

// ---------------------------------------------------------------------------
// fp32 -> bf16 conversion (grid-stride)
// ---------------------------------------------------------------------------
__global__ void cvt_f32_bf16_kernel(const float* __restrict__ in,
                                    bf16_t* __restrict__ out, int n) {
  int i = blockIdx.x * blockDim.x + threadIdx.x;
  int stride = gridDim.x * blockDim.x;
  for (; i < n; i += stride) out[i] = (bf16_t)in[i];
}

// ---------------------------------------------------------------------------
// QKV projection: C[m,n] = x_bf[m,:] . Wqkv_bf[n,:] + b_qkv[n]
// m in [0,4096)  (b = m/T, t = m%T),  n in [0,3072)
// Scatter: Q,K -> [B,H,T,D] bf16,  V -> [B,H,D,T] bf16 (transposed)
// Block: 256 thr = 8 waves (2x4), block tile 64(M) x 128(N), wave tile 32x32
// ---------------------------------------------------------------------------
__global__ __launch_bounds__(256) void qkv_gemm_kernel(
    const bf16_t* __restrict__ A, const bf16_t* __restrict__ W,
    const float* __restrict__ bias,
    bf16_t* __restrict__ Qb, bf16_t* __restrict__ Kb, bf16_t* __restrict__ Vt) {
  const int lane = threadIdx.x & 31;
  const int wave = threadIdx.x >> 5;
  const int wm = wave & 1, wn = wave >> 1;
  const int Mbase = blockIdx.y * 64 + wm * 32;
  const int Nbase = blockIdx.x * 128 + wn * 32;

  v8f c[2][2] = {};
  for (int k = 0; k < DIM; k += 32) {
    v16bf a0 = load_frag_a(A, DIM, Mbase,      k, lane);
    v16bf a1 = load_frag_a(A, DIM, Mbase + 16, k, lane);
    v16bf b0 = load_frag_b(W, DIM, Nbase,      k, lane);
    v16bf b1 = load_frag_b(W, DIM, Nbase + 16, k, lane);
    c[0][0] = wmma_bf16(a0, b0, c[0][0]);
    c[0][1] = wmma_bf16(a0, b1, c[0][1]);
    c[1][0] = wmma_bf16(a1, b0, c[1][0]);
    c[1][1] = wmma_bf16(a1, b1, c[1][1]);
  }

  const int hi = lane >> 4, lo = lane & 15;
#pragma unroll
  for (int i = 0; i < 2; i++)
#pragma unroll
    for (int j = 0; j < 2; j++) {
      const int n = Nbase + j * 16 + lo;
      const float bn = bias[n];
      const int which = n >> 10;           // 0=Q 1=K 2=V (uniform per wave)
      const int rem = n & (DIM - 1);
      const int h = rem >> 6, d = rem & 63;
#pragma unroll
      for (int r = 0; r < 8; r++) {
        const int m = Mbase + i * 16 + r + 8 * hi;
        const int b = m >> 11, t = m & (TSEQ - 1);
        const bf16_t val = (bf16_t)(c[i][j][r] + bn);
        const size_t head = (size_t)(b * NHEAD + h);
        if (which == 0)      Qb[(head * TSEQ + t) * HDIM + d] = val;
        else if (which == 1) Kb[(head * TSEQ + t) * HDIM + d] = val;
        else                 Vt[(head * HDIM + d) * TSEQ + t] = val;
      }
    }
}

// ---------------------------------------------------------------------------
// Flash attention: grid (T/64, H, B), block 128 thr = 4 waves.
// Wave w owns 16 query rows. Online softmax over 64-key tiles.
// AO written as [B,T,H*D] bf16 (ready for the output GEMM).
// ---------------------------------------------------------------------------
__global__ __launch_bounds__(128) void flash_attn_kernel(
    const bf16_t* __restrict__ Qb, const bf16_t* __restrict__ Kb,
    const bf16_t* __restrict__ Vt, const float* __restrict__ cutoffs,
    bf16_t* __restrict__ AO) {
  const int lane = threadIdx.x & 31;
  const int w    = threadIdx.x >> 5;
  const int h    = blockIdx.y;
  const int b    = blockIdx.z;
  const int qbase = blockIdx.x * 64 + w * 16;
  const int hi = lane >> 4, lo = lane & 15;

  const size_t headoff = (size_t)(b * NHEAD + h) * TSEQ * HDIM;
  const bf16_t* Qh = Qb + headoff;
  const bf16_t* Kh = Kb + headoff;
  const bf16_t* Vh = Vt + headoff;                 // [D,T] for this head
  const float*  Ch = cutoffs + (size_t)b * TSEQ * TSEQ;

  __shared__ bf16_t plds[4][16][64];               // per-wave P staging (8 KB)

  const v16bf qa0 = load_frag_a(Qh, HDIM, qbase, 0,  lane);
  const v16bf qa1 = load_frag_a(Qh, HDIM, qbase, 32, lane);

  v8f o[4] = {};
  float m8[8], l8[8];
#pragma unroll
  for (int r = 0; r < 8; r++) { m8[r] = -3.0e38f; l8[r] = 0.0f; }

  for (int kt = 0; kt < TSEQ / 64; kt++) {
    const int kbase = kt * 64;

    // ---- S = Q @ K^T  (16 x 64 tile) ----
    v8f s[4];
#pragma unroll
    for (int nt = 0; nt < 4; nt++) {
      v16bf kb0 = load_frag_b(Kh, HDIM, kbase + nt * 16, 0,  lane);
      v16bf kb1 = load_frag_b(Kh, HDIM, kbase + nt * 16, 32, lane);
      v8f acc = {};
      acc = wmma_bf16(qa0, kb0, acc);
      acc = wmma_bf16(qa1, kb1, acc);
      s[nt] = acc;
    }

    // ---- scale + additive log-bias ----
#pragma unroll
    for (int nt = 0; nt < 4; nt++) {
      const int kcol = kbase + nt * 16 + lo;
#pragma unroll
      for (int r = 0; r < 8; r++) {
        const int q = qbase + r + 8 * hi;
        const float cv = Ch[(size_t)q * TSEQ + kcol];
        s[nt][r] = s[nt][r] * SCALE + __logf(fmaxf(cv, MINLOG));
      }
    }

    // ---- online softmax (row reductions across 16-lane halves) ----
#pragma unroll
    for (int r = 0; r < 8; r++) {
      float mx = fmaxf(fmaxf(s[0][r], s[1][r]), fmaxf(s[2][r], s[3][r]));
      mx = fmaxf(mx, __shfl_xor(mx, 1, 32));
      mx = fmaxf(mx, __shfl_xor(mx, 2, 32));
      mx = fmaxf(mx, __shfl_xor(mx, 4, 32));
      mx = fmaxf(mx, __shfl_xor(mx, 8, 32));
      const float mnew = fmaxf(m8[r], mx);
      const float sc = __expf(m8[r] - mnew);
      m8[r] = mnew;
      float rs = 0.0f;
#pragma unroll
      for (int nt = 0; nt < 4; nt++) {
        const float pv = __expf(s[nt][r] - mnew);
        s[nt][r] = pv;
        rs += pv;
      }
      rs += __shfl_xor(rs, 1, 32);
      rs += __shfl_xor(rs, 2, 32);
      rs += __shfl_xor(rs, 4, 32);
      rs += __shfl_xor(rs, 8, 32);
      l8[r] = l8[r] * sc + rs;
#pragma unroll
      for (int dt = 0; dt < 4; dt++) o[dt][r] *= sc;
    }

    // ---- P: C-layout -> bf16 A-layout via per-wave LDS ----
#pragma unroll
    for (int nt = 0; nt < 4; nt++)
#pragma unroll
      for (int r = 0; r < 8; r++)
        plds[w][r + 8 * hi][nt * 16 + lo] = (bf16_t)s[nt][r];

    const v16bf pa0 = load_frag_a(&plds[w][0][0], 64, 0, 0,  lane);
    const v16bf pa1 = load_frag_a(&plds[w][0][0], 64, 0, 32, lane);

    // ---- O += P @ V ----
#pragma unroll
    for (int dt = 0; dt < 4; dt++) {
      v16bf vb0 = load_frag_b(Vh, TSEQ, dt * 16, kbase,      lane);
      v16bf vb1 = load_frag_b(Vh, TSEQ, dt * 16, kbase + 32, lane);
      o[dt] = wmma_bf16(pa0, vb0, o[dt]);
      o[dt] = wmma_bf16(pa1, vb1, o[dt]);
    }
  }

  // ---- normalize + store [B,T,H*D] bf16 ----
#pragma unroll
  for (int r = 0; r < 8; r++) {
    const float inv = 1.0f / l8[r];
    const int q = qbase + r + 8 * hi;
#pragma unroll
    for (int dt = 0; dt < 4; dt++) {
      AO[((size_t)(b * TSEQ + q)) * DIM + h * HDIM + dt * 16 + lo] =
          (bf16_t)(o[dt][r] * inv);
    }
  }
}

// ---------------------------------------------------------------------------
// Output projection: out[m,n] = AO[m,:] . Wout_bf[n,:] + b_out[n]  (fp32 out)
// ---------------------------------------------------------------------------
__global__ __launch_bounds__(256) void out_gemm_kernel(
    const bf16_t* __restrict__ A, const bf16_t* __restrict__ W,
    const float* __restrict__ bias, float* __restrict__ Out) {
  const int lane = threadIdx.x & 31;
  const int wave = threadIdx.x >> 5;
  const int wm = wave & 1, wn = wave >> 1;
  const int Mbase = blockIdx.y * 64 + wm * 32;
  const int Nbase = blockIdx.x * 128 + wn * 32;

  v8f c[2][2] = {};
  for (int k = 0; k < DIM; k += 32) {
    v16bf a0 = load_frag_a(A, DIM, Mbase,      k, lane);
    v16bf a1 = load_frag_a(A, DIM, Mbase + 16, k, lane);
    v16bf b0 = load_frag_b(W, DIM, Nbase,      k, lane);
    v16bf b1 = load_frag_b(W, DIM, Nbase + 16, k, lane);
    c[0][0] = wmma_bf16(a0, b0, c[0][0]);
    c[0][1] = wmma_bf16(a0, b1, c[0][1]);
    c[1][0] = wmma_bf16(a1, b0, c[1][0]);
    c[1][1] = wmma_bf16(a1, b1, c[1][1]);
  }

  const int hi = lane >> 4, lo = lane & 15;
#pragma unroll
  for (int i = 0; i < 2; i++)
#pragma unroll
    for (int j = 0; j < 2; j++) {
      const int n = Nbase + j * 16 + lo;
      const float bn = bias[n];
#pragma unroll
      for (int r = 0; r < 8; r++) {
        const int m = Mbase + i * 16 + r + 8 * hi;
        Out[(size_t)m * DIM + n] = c[i][j][r] + bn;
      }
    }
}

// ---------------------------------------------------------------------------
// Launcher
// ---------------------------------------------------------------------------
extern "C" void kernel_launch(void* const* d_in, const int* in_sizes, int n_in,
                              void* d_out, int out_size, void* d_ws, size_t ws_size,
                              hipStream_t stream) {
  const float* x      = (const float*)d_in[0];   // [B,T,DIM]
  const float* cuts   = (const float*)d_in[1];   // [B,T,T]
  const float* W_qkv  = (const float*)d_in[2];   // [3*DIM, DIM]
  const float* b_qkv  = (const float*)d_in[3];   // [3*DIM]
  const float* W_out  = (const float*)d_in[4];   // [DIM, DIM]
  const float* b_out  = (const float*)d_in[5];   // [DIM]
  float* out = (float*)d_out;

  char* ws = (char*)d_ws;
  const size_t MiB = 1u << 20;
  bf16_t* x_bf    = (bf16_t*)(ws + 0 * MiB);    // 8 MiB
  bf16_t* wqkv_bf = (bf16_t*)(ws + 8 * MiB);    // 6 MiB
  bf16_t* wout_bf = (bf16_t*)(ws + 14 * MiB);   // 2 MiB
  bf16_t* Qb      = (bf16_t*)(ws + 16 * MiB);   // 8 MiB [B,H,T,D]
  bf16_t* Kb      = (bf16_t*)(ws + 24 * MiB);   // 8 MiB [B,H,T,D]
  bf16_t* Vt      = (bf16_t*)(ws + 32 * MiB);   // 8 MiB [B,H,D,T]
  bf16_t* AO      = (bf16_t*)(ws + 40 * MiB);   // 8 MiB [B,T,DIM]

  const int n_x    = BATCH * TSEQ * DIM;   // 4,194,304
  const int n_wqkv = 3 * DIM * DIM;        // 3,145,728
  const int n_wout = DIM * DIM;            // 1,048,576

  cvt_f32_bf16_kernel<<<2048, 256, 0, stream>>>(x, x_bf, n_x);
  cvt_f32_bf16_kernel<<<2048, 256, 0, stream>>>(W_qkv, wqkv_bf, n_wqkv);
  cvt_f32_bf16_kernel<<<2048, 256, 0, stream>>>(W_out, wout_bf, n_wout);

  qkv_gemm_kernel<<<dim3(3 * DIM / 128, BATCH * TSEQ / 64), 256, 0, stream>>>(
      x_bf, wqkv_bf, b_qkv, Qb, Kb, Vt);

  flash_attn_kernel<<<dim3(TSEQ / 64, NHEAD, BATCH), 128, 0, stream>>>(
      Qb, Kb, Vt, cuts, AO);

  out_gemm_kernel<<<dim3(DIM / 128, BATCH * TSEQ / 64), 256, 0, stream>>>(
      AO, wout_bf, b_out, out);
}
